// MaskRcnn_14439680049751
// MI455X (gfx1250) — matrix operand order repaired and checked by reading, not compiled, not run on hardware
//
#include <hip/hip_runtime.h>
#include <hip/hip_bf16.h>
#include <math.h>

typedef __attribute__((ext_vector_type(16))) _Float16 v16h;
typedef __attribute__((ext_vector_type(8)))  float    v8f;
typedef __attribute__((ext_vector_type(4)))  float    v4f;

#define BN_EPS 1e-3f

// ---------------------------------------------------------------------------
// Fragment helpers (wave32 WMMA 16x16x32 f16 layouts).
//   A frag, lane half h: k = e + 16h         -> one contiguous 16-float run
//   B frag, lane half h: k = e + 16h         -> one contiguous 16-float run
// All K dims used here are multiples of 32 and rows are 16B aligned, so the
// runs load as global_load_b128.
// ---------------------------------------------------------------------------
__device__ __forceinline__ v16h pack16(v4f x0, v4f x1, v4f x2, v4f x3)
{
    v16h r;
    r[0]=(_Float16)x0[0];  r[1]=(_Float16)x0[1];  r[2]=(_Float16)x0[2];  r[3]=(_Float16)x0[3];
    r[4]=(_Float16)x1[0];  r[5]=(_Float16)x1[1];  r[6]=(_Float16)x1[2];  r[7]=(_Float16)x1[3];
    r[8]=(_Float16)x2[0];  r[9]=(_Float16)x2[1];  r[10]=(_Float16)x2[2]; r[11]=(_Float16)x2[3];
    r[12]=(_Float16)x3[0]; r[13]=(_Float16)x3[1]; r[14]=(_Float16)x3[2]; r[15]=(_Float16)x3[3];
    return r;
}
__device__ __forceinline__ v16h zfrag()
{
    v16h r;
#pragma unroll
    for (int e = 0; e < 16; ++e) r[e] = (_Float16)0.f;
    return r;
}
__device__ __forceinline__ v16h loadA(const float* row, int k0, int half)
{
    const float* p0 = row + k0 + half * 16;
    const float* p1 = p0 + 8;
    return pack16(*(const v4f*)p0, *(const v4f*)(p0 + 4),
                  *(const v4f*)p1, *(const v4f*)(p1 + 4));
}
__device__ __forceinline__ v16h loadB16(const float* row, int k0, int half)
{
    const float* p = row + k0 + half * 16;
    return pack16(*(const v4f*)p, *(const v4f*)(p + 4),
                  *(const v4f*)(p + 8), *(const v4f*)(p + 12));
}
#define WMMA_F16(A, B, C) \
    __builtin_amdgcn_wmma_f32_16x16x32_f16(false, A, false, B, (short)0, C, false, false)

// ============================================================================
// Conv (stride1 "same") as implicit GEMM, 2x2 wave tiling (32 OC x 32 pos).
//   in : (Nimg, IC, H, W)   wgt: (OC, IC, KH, KH)   out: (Nimg, OC, H, W)
//   epi: 0 +bias | 1 +bias,relu | 2 +bias,bn,relu | 3 +bias,sigmoid
// Requires IC*KH*KH % 32 == 0 (true for all uses: 2304, 512, 256).
// ============================================================================
template <int KH>
__global__ void conv_wmma_kernel(const float* __restrict__ in,
                                 const float* __restrict__ wgt,
                                 const float* __restrict__ bias,
                                 const float* __restrict__ bng, const float* __restrict__ bnb,
                                 const float* __restrict__ bnm, const float* __restrict__ bnv,
                                 float* __restrict__ out,
                                 int Nimg, int IC, int H, int W, int OC,
                                 int pad, int epi)
{
    const int KW = KH, KK = KH * KH;
    const int lane = threadIdx.x & 31;
    const int wid  = threadIdx.x >> 5;
    const int HW = H * W;
    const int P  = Nimg * HW;
    const int pT2 = (P + 31) >> 5;
    const int oT2 = (OC + 31) >> 5;
    int tile = blockIdx.x * (blockDim.x >> 5) + wid;
    if (tile >= pT2 * oT2) return;
    const int oT = tile / pT2;
    const int pT = tile - oT * pT2;

    const int half = lane >> 4;
    const int l15  = lane & 15;
    const int K = IC * KK;

    const int m0 = oT * 32 + l15, m1 = m0 + 16;
    const int pos0 = pT * 32 + l15, pos1 = pos0 + 16;
    int b0 = pos0 / HW, rem0 = pos0 - b0 * HW;
    int oy0 = rem0 / W, ox0 = rem0 - oy0 * W;
    int b1 = pos1 / HW, rem1 = pos1 - b1 * HW;
    int oy1 = rem1 / W, ox1 = rem1 - oy1 * W;
    const bool pv0 = pos0 < P, pv1 = pos1 < P;
    const bool av0 = m0 < OC, av1 = m1 < OC;
    const float* w0r = wgt + (size_t)m0 * K;
    const float* w1r = wgt + (size_t)m1 * K;

    v8f c00 = {}, c01 = {}, c10 = {}, c11 = {};
    for (int k0 = 0; k0 < K; k0 += 32) {
        // speculative prefetch ahead in K (ISA 10.5: dropped on bad address,
        // no counter, no fault) -> no guards, no exec-mask churn
        __builtin_prefetch(w0r + k0 + 128, 0, 1);
        __builtin_prefetch(w1r + k0 + 128, 0, 1);
        v16h a0 = av0 ? loadA(w0r, k0, half) : zfrag();
        v16h a1 = av1 ? loadA(w1r, k0, half) : zfrag();
        v16h bb0, bb1;
#pragma unroll
        for (int e = 0; e < 16; ++e) {
            int kb = k0 + e + half * 16;
            float f0 = 0.f, f1 = 0.f;
            if (KH == 1) {
                if (pv0) f0 = in[((size_t)b0 * IC + kb) * HW + rem0];
                if (pv1) f1 = in[((size_t)b1 * IC + kb) * HW + rem1];
            } else {
                int ic = kb / KK; int r = kb - ic * KK;
                int ky = r / KW;  int kx = r - ky * KW;
                int iy0 = oy0 + ky - pad, ix0 = ox0 + kx - pad;
                int iy1 = oy1 + ky - pad, ix1 = ox1 + kx - pad;
                if (pv0 && iy0 >= 0 && iy0 < H && ix0 >= 0 && ix0 < W)
                    f0 = in[(((size_t)b0 * IC + ic) * H + iy0) * W + ix0];
                if (pv1 && iy1 >= 0 && iy1 < H && ix1 >= 0 && ix1 < W)
                    f1 = in[(((size_t)b1 * IC + ic) * H + iy1) * W + ix1];
            }
            bb0[e] = (_Float16)f0;
            bb1[e] = (_Float16)f1;
        }
        c00 = WMMA_F16(a0, bb0, c00);
        c01 = WMMA_F16(a0, bb1, c01);
        c10 = WMMA_F16(a1, bb0, c10);
        c11 = WMMA_F16(a1, bb1, c11);
    }

    auto store = [&](v8f& c, int mBase, bool pvalid, int bb, int oy, int ox) {
        if (!pvalid) return;
#pragma unroll
        for (int r = 0; r < 8; ++r) {
            int mm = mBase + r + half * 8;
            if (mm >= OC) continue;
            float v = c[r] + bias[mm];
            if (epi == 2) {
                float s = bng[mm] * rsqrtf(bnv[mm] + BN_EPS);
                v = (v - bnm[mm]) * s + bnb[mm];
            }
            if (epi == 1 || epi == 2) v = fmaxf(v, 0.f);
            if (epi == 3) v = 1.f / (1.f + expf(-v));
            out[(((size_t)bb * OC + mm) * H + oy) * W + ox] = v;
        }
    };
    store(c00, oT * 32,      pv0, b0, oy0, ox0);
    store(c01, oT * 32,      pv1, b1, oy1, ox1);
    store(c10, oT * 32 + 16, pv0, b0, oy0, ox0);
    store(c11, oT * 32 + 16, pv1, b1, oy1, ox1);
}

// ============================================================================
// Dense GEMM: out[M,N] = X[M,K] * Wm[N,K]^T (+bias over N, optional BN over N)
//   epi: 0 bias | 2 bias+bn+relu | 4 bias+bn.   K % 32 == 0.
// 2x2 wave tiling, all-b128 fragment loads, unconditional speculative prefetch.
// ============================================================================
__global__ void gemm_wmma_kernel(const float* __restrict__ X,
                                 const float* __restrict__ Wm,
                                 const float* __restrict__ bias,
                                 const float* __restrict__ bng, const float* __restrict__ bnb,
                                 const float* __restrict__ bnm, const float* __restrict__ bnv,
                                 float* __restrict__ out, int M, int N, int K, int epi)
{
    const int lane = threadIdx.x & 31;
    const int wid  = threadIdx.x >> 5;
    const int mT2 = (M + 31) >> 5, nT2 = (N + 31) >> 5;
    int tile = blockIdx.x * (blockDim.x >> 5) + wid;
    if (tile >= mT2 * nT2) return;
    const int nT = tile / mT2;
    const int mT = tile - nT * mT2;
    const int half = lane >> 4, l15 = lane & 15;
    const int mr0 = mT * 32 + l15, mr1 = mr0 + 16;
    const int nc0 = nT * 32 + l15, nc1 = nc0 + 16;
    const bool am0 = mr0 < M, am1 = mr1 < M, bv0 = nc0 < N, bv1 = nc1 < N;
    const float* xr0 = X + (size_t)mr0 * K;
    const float* xr1 = X + (size_t)mr1 * K;
    const float* wr0 = Wm + (size_t)nc0 * K;
    const float* wr1 = Wm + (size_t)nc1 * K;

    v8f c00 = {}, c01 = {}, c10 = {}, c11 = {};
    for (int k0 = 0; k0 < K; k0 += 32) {
        __builtin_prefetch(xr0 + k0 + 128, 0, 1);   // speculative, guard-free
        __builtin_prefetch(wr0 + k0 + 128, 0, 1);
        v16h a0 = am0 ? loadA(xr0, k0, half) : zfrag();
        v16h a1 = am1 ? loadA(xr1, k0, half) : zfrag();
        v16h b0 = bv0 ? loadB16(wr0, k0, half) : zfrag();
        v16h b1 = bv1 ? loadB16(wr1, k0, half) : zfrag();
        c00 = WMMA_F16(a0, b0, c00);
        c01 = WMMA_F16(a0, b1, c01);
        c10 = WMMA_F16(a1, b0, c10);
        c11 = WMMA_F16(a1, b1, c11);
    }

    auto store = [&](v8f& c, int mBase, int nn) {
        if (nn >= N) return;
#pragma unroll
        for (int r = 0; r < 8; ++r) {
            int mm = mBase + r + half * 8;
            if (mm >= M) continue;
            float v = c[r] + bias[nn];
            if (epi == 2 || epi == 4) {
                float s = bng[nn] * rsqrtf(bnv[nn] + BN_EPS);
                v = (v - bnm[nn]) * s + bnb[nn];
            }
            if (epi == 2) v = fmaxf(v, 0.f);
            out[(size_t)mm * N + nn] = v;
        }
    };
    store(c00, mT * 32,      nc0);
    store(c01, mT * 32,      nc1);
    store(c10, mT * 32 + 16, nc0);
    store(c11, mT * 32 + 16, nc1);
}

// ============================================================================
// 2x2 stride-2 deconv: out[n,o,2y+ay,2x+ax] = relu(sum_c h[n,c,y,x]*wd[c,o,ay,ax]+bd)
//   h: (Nimg,256,14,14)  wd: (256,256,2,2)  out: (Nimg,256,28,28)
// Four GEMMs (one per (ay,ax)); 2x2 wave tiling.
// ============================================================================
__global__ void deconv_wmma_kernel(const float* __restrict__ h,
                                   const float* __restrict__ wd,
                                   const float* __restrict__ bd,
                                   float* __restrict__ out, int Nimg)
{
    const int lane = threadIdx.x & 31;
    const int wid  = threadIdx.x >> 5;
    const int P = Nimg * 196;
    const int pT2 = (P + 31) >> 5;
    int tile = blockIdx.x * (blockDim.x >> 5) + wid;
    if (tile >= pT2 * 8 * 4) return;
    const int ab = tile / (pT2 * 8);
    int rem = tile - ab * pT2 * 8;
    const int oT = rem / pT2;
    const int pT = rem - oT * pT2;
    const int ay = ab >> 1, ax = ab & 1;
    const int half = lane >> 4, l15 = lane & 15;
    const int m0 = oT * 32 + l15, m1 = m0 + 16;
    const int pos0 = pT * 32 + l15, pos1 = pos0 + 16;
    int b0 = pos0 / 196, r20 = pos0 - b0 * 196;
    int y0 = r20 / 14, x0 = r20 - 14 * y0;
    int b1 = pos1 / 196, r21 = pos1 - b1 * 196;
    int y1 = r21 / 14, x1 = r21 - 14 * y1;
    const bool pv0 = pos0 < P, pv1 = pos1 < P;

    v8f c00 = {}, c01 = {}, c10 = {}, c11 = {};
    for (int k0 = 0; k0 < 256; k0 += 32) {
        v16h a0, a1, bb0, bb1;
#pragma unroll
        for (int e = 0; e < 16; ++e) {
            int ka = k0 + e + half * 16;
            a0[e] = (_Float16)wd[(((size_t)ka * 256 + m0) * 2 + ay) * 2 + ax];
            a1[e] = (_Float16)wd[(((size_t)ka * 256 + m1) * 2 + ay) * 2 + ax];
            int kb = k0 + e + half * 16;
            bb0[e] = (_Float16)(pv0 ? h[((size_t)b0 * 256 + kb) * 196 + r20] : 0.f);
            bb1[e] = (_Float16)(pv1 ? h[((size_t)b1 * 256 + kb) * 196 + r21] : 0.f);
        }
        c00 = WMMA_F16(a0, bb0, c00);
        c01 = WMMA_F16(a0, bb1, c01);
        c10 = WMMA_F16(a1, bb0, c10);
        c11 = WMMA_F16(a1, bb1, c11);
    }

    auto store = [&](v8f& c, int mBase, bool pv, int bb, int yy, int xx) {
        if (!pv) return;
#pragma unroll
        for (int r = 0; r < 8; ++r) {
            int mm = mBase + r + half * 8;
            float v = fmaxf(c[r] + bd[mm], 0.f);
            out[(((size_t)bb * 256 + mm) * 28 + (2 * yy + ay)) * 28 + (2 * xx + ax)] = v;
        }
    };
    store(c00, oT * 32,      pv0, b0, y0, x0);
    store(c01, oT * 32,      pv1, b1, y1, x1);
    store(c10, oT * 32 + 16, pv0, b0, y0, x0);
    store(c11, oT * 32 + 16, pv1, b1, y1, x1);
}

// ============================================================================
// Proposal generation (one block per image): sigmoid score + exact 50-pass
// argmax top-k (lax.top_k tie semantics) + box decode with the reference's
// buggy anchor-base overwrite loop.
// ============================================================================
__device__ __forceinline__ void anchor_base(int a, float stride,
                                            float& x1, float& y1, float& x2, float& y2)
{
    const float scales[5] = {4.f, 8.f, 16.f, 32.f, 64.f};
    const float ratios[3] = {0.5f, 1.f, 2.f};
    float p = stride * 0.5f;
    x1 = y1 = x2 = y2 = 0.f;                 // base[11..14] stay zero
    for (int i = 0; i < 3; ++i)
        for (int j = 0; j < 5; ++j)
            if (i * 3 + j == a) {            // replicate bId collision (last wins)
                float w = stride * scales[j] * sqrtf(ratios[i]);
                float h = stride * scales[j] / sqrtf(ratios[i]);
                x1 = p - w * 0.5f; y1 = p - h * 0.5f;
                x2 = p + w * 0.5f; y2 = p + h * 0.5f;
            }
}

__global__ void proposals_kernel(const float* __restrict__ cls,
                                 const float* __restrict__ bbox,
                                 float* __restrict__ sc,
                                 float* __restrict__ boxes,  // (5,2,50,4)
                                 int level, int Hf, float stride)
{
    const int b = blockIdx.x;
    const int N = Hf * Hf * 15;
    float* s = sc + (size_t)b * 162240;
    for (int j = threadIdx.x; j < N; j += blockDim.x) {
        int a = j % 15, p = j / 15;
        int y = p / Hf, x = p - y * Hf;
        float c0 = cls[(((size_t)b * 30 + 2 * a)     * Hf + y) * Hf + x];
        float c1 = cls[(((size_t)b * 30 + 2 * a + 1) * Hf + y) * Hf + x];
        s[j] = 1.f / (1.f + expf(-(c1 - c0)));
    }
    __syncthreads();

    __shared__ unsigned long long red[256];
    for (int it = 0; it < 50; ++it) {
        unsigned long long best = 0ull;
        for (int j = threadIdx.x; j < N; j += blockDim.x) {
            unsigned u = __float_as_uint(s[j]);
            u = (u & 0x80000000u) ? ~u : (u | 0x80000000u);
            unsigned long long key = ((unsigned long long)u << 32) | (unsigned)(~j);
            if (key > best) best = key;
        }
        red[threadIdx.x] = best;
        __syncthreads();
        for (int st = blockDim.x >> 1; st > 0; st >>= 1) {
            if ((int)threadIdx.x < st)
                red[threadIdx.x] = red[threadIdx.x] > red[threadIdx.x + st]
                                       ? red[threadIdx.x] : red[threadIdx.x + st];
            __syncthreads();
        }
        if (threadIdx.x == 0) {
            int idx = (int)(~((unsigned)(red[0] & 0xFFFFFFFFull)));
            int a = idx % 15, p = idx / 15;
            int hy = p / Hf, wx = p - hy * Hf;
            float ax1, ay1, ax2, ay2;
            anchor_base(a, stride, ax1, ay1, ax2, ay2);
            float x1 = wx * stride + ax1, y1 = hy * stride + ay1;
            float x2 = wx * stride + ax2, y2 = hy * stride + ay2;
            float ww = x2 - x1, hh = y2 - y1;
            float cx = x1 + ww * 0.5f, cy = y1 + hh * 0.5f;
            float d0 = bbox[(((size_t)b * 60 + a * 4 + 0) * Hf + hy) * Hf + wx];
            float d1 = bbox[(((size_t)b * 60 + a * 4 + 1) * Hf + hy) * Hf + wx];
            float d2 = bbox[(((size_t)b * 60 + a * 4 + 2) * Hf + hy) * Hf + wx];
            float d3 = bbox[(((size_t)b * 60 + a * 4 + 3) * Hf + hy) * Hf + wx];
            float gx = cx + ww * d0, gy = cy + hh * d1;
            float gw = ww * expf(d2), gh = hh * expf(d3);
            float bx1 = fminf(fmaxf(gx - gw * 0.5f, 0.f), (float)Hf);
            float by1 = fminf(fmaxf(gy - gh * 0.5f, 0.f), (float)Hf);
            float* dst = boxes + (((size_t)(level * 2 + b) * 50 + it) * 4);
            dst[0] = bx1; dst[1] = by1; dst[2] = 1.5f * gw; dst[3] = 1.5f * gh;
            s[idx] = -1.f;
        }
        __syncthreads();
    }
}

// ============================================================================
// ROI align, S=2 sampling, faithful to reference (boxes stored (x1,y1,w,h) but
// consumed as (x1,y1,x2,y2)). out: (B,250,C,P,P)
// ============================================================================
__global__ void roi_align_kernel(const float* __restrict__ f0, const float* __restrict__ f1,
                                 const float* __restrict__ f2, const float* __restrict__ f3,
                                 const float* __restrict__ f4,
                                 const float* __restrict__ boxes,
                                 float* __restrict__ out, int P)
{
    const int C = 256;
    int t = blockIdx.x * blockDim.x + threadIdx.x;
    int total = 2 * 250 * C * P * P;
    if (t >= total) return;
    int px = t % P;  int rem = t / P;
    int py = rem % P; rem /= P;
    int c  = rem % C; rem /= C;
    int local = rem % 250;  int b = rem / 250;
    int level = local / 50; int box = local % 50;

    const int   fs[5]  = {104, 52, 26, 13, 7};
    const float isc[5] = {0.25f, 0.125f, 0.0625f, 0.03125f, 0.015625f};
    const float* fp[5] = {f0, f1, f2, f3, f4};
    int FH = fs[level];
    const float* feat = fp[level] + ((size_t)b * C + c) * FH * FH;
    const float* bx = boxes + (((size_t)(level * 2 + b) * 50 + box) * 4);
    float scale = isc[level];
    float x1 = bx[0] * scale, y1 = bx[1] * scale;
    float x2 = bx[2] * scale, y2 = bx[3] * scale;
    float rw = fmaxf(x2 - x1, 1.f), rh = fmaxf(y2 - y1, 1.f);
    float bw = rw / P, bh = rh / P;
    const float g[2] = {0.25f, 0.75f};

    float acc = 0.f;
#pragma unroll
    for (int sy = 0; sy < 2; ++sy)
#pragma unroll
        for (int sx = 0; sx < 2; ++sx) {
            float ys = y1 + py * bh + g[sy] * bh;
            float xs = x1 + px * bw + g[sx] * bw;
            bool vy = (ys > -1.f) && (ys < (float)FH);
            bool vx = (xs > -1.f) && (xs < (float)FH);
            float yc = fminf(fmaxf(ys, 0.f), FH - 1.f);
            float xc = fminf(fmaxf(xs, 0.f), FH - 1.f);
            int y0 = (int)floorf(yc), x0 = (int)floorf(xc);
            int y1i = min(y0 + 1, FH - 1), x1i = min(x0 + 1, FH - 1);
            float ly = yc - y0, lx = xc - x0;
            float f00 = feat[y0 * FH + x0],  f01 = feat[y0 * FH + x1i];
            float f10 = feat[y1i * FH + x0], f11 = feat[y1i * FH + x1i];
            float v = f00 * (1 - ly) * (1 - lx) + f01 * (1 - ly) * lx
                    + f10 * ly * (1 - lx)       + f11 * ly * lx;
            acc += (vy && vx) ? v : 0.f;
        }
    out[(((size_t)(b * 250 + local) * C + c) * P + py) * P + px] = acc * 0.25f;
}

// NCHW -> (B, H*W*A, CC) gather for the raw RPN level-4 outputs.
__global__ void rpn_head_transpose_kernel(const float* __restrict__ src,
                                          float* __restrict__ dst, int CC, int Hf)
{
    int t = blockIdx.x * blockDim.x + threadIdx.x;
    int tot = 2 * Hf * Hf * 15 * CC;
    if (t >= tot) return;
    int cc = t % CC; int r = t / CC;
    int a = r % 15;  r /= 15;
    int x = r % Hf;  r /= Hf;
    int y = r % Hf;  int b = r / Hf;
    dst[t] = src[(((size_t)b * (15 * CC) + a * CC + cc) * Hf + y) * Hf + x];
}

// ============================================================================
extern "C" void kernel_launch(void* const* d_in, const int* in_sizes, int n_in,
                              void* d_out, int out_size, void* d_ws, size_t ws_size,
                              hipStream_t stream)
{
    const float* feat[5];
    for (int i = 0; i < 5; ++i) feat[i] = (const float*)d_in[i];
    // params: JAX pytree (alphabetical) flatten order: cls, mask, rpn
    const float* cls_b1  = (const float*)d_in[5];
    const float* cls_b2  = (const float*)d_in[6];
    const float* cls_bbb = (const float*)d_in[7];
    const float* cls_bfc = (const float*)d_in[8];
    const float* bn1_b = (const float*)d_in[9],  *bn1_g = (const float*)d_in[10];
    const float* bn1_m = (const float*)d_in[11], *bn1_v = (const float*)d_in[12];
    const float* bn2_b = (const float*)d_in[13], *bn2_g = (const float*)d_in[14];
    const float* bn2_m = (const float*)d_in[15], *bn2_v = (const float*)d_in[16];
    const float* cls_w1  = (const float*)d_in[17];
    const float* cls_w2  = (const float*)d_in[18];
    const float* cls_wbb = (const float*)d_in[19];
    const float* cls_wfc = (const float*)d_in[20];
    const float* m_b[4]  = {(const float*)d_in[21], (const float*)d_in[22],
                            (const float*)d_in[23], (const float*)d_in[24]};
    const float* m_bd = (const float*)d_in[25];
    const float* m_bf = (const float*)d_in[26];
    const float* m_bn_b[4], *m_bn_g[4], *m_bn_m[4], *m_bn_v[4];
    for (int i = 0; i < 4; ++i) {
        m_bn_b[i] = (const float*)d_in[27 + 4 * i + 0];
        m_bn_g[i] = (const float*)d_in[27 + 4 * i + 1];
        m_bn_m[i] = (const float*)d_in[27 + 4 * i + 2];
        m_bn_v[i] = (const float*)d_in[27 + 4 * i + 3];
    }
    const float* m_w[4]  = {(const float*)d_in[43], (const float*)d_in[44],
                            (const float*)d_in[45], (const float*)d_in[46]};
    const float* m_wd = (const float*)d_in[47];
    const float* m_wf = (const float*)d_in[48];
    const float* rpn_bb = (const float*)d_in[49];
    const float* rpn_bc = (const float*)d_in[50];
    const float* rpn_bs = (const float*)d_in[51];
    const float* rpn_wb = (const float*)d_in[52];
    const float* rpn_wc = (const float*)d_in[53];
    const float* rpn_ws = (const float*)d_in[54];

    // workspace layout (floats)
    float* ws = (float*)d_ws;
    size_t o = 0;
    float* h_rpn   = ws + o; o += (size_t)2 * 512 * 104 * 104;
    float* clsBuf  = ws + o; o += (size_t)2 * 30 * 104 * 104;
    float* bboxBuf = ws + o; o += (size_t)2 * 60 * 104 * 104;
    float* scoreB  = ws + o; o += (size_t)2 * 162240;
    float* boxes   = ws + o; o += 5 * 2 * 50 * 4;
    float* pf      = ws + o; o += (size_t)2 * 250 * 256 * 49;
    float* mfb     = ws + o; o += (size_t)2 * 250 * 256 * 196;
    float* h1      = ws + o; o += (size_t)500 * 1024;
    float* h2      = ws + o; o += (size_t)500 * 1024;
    float* mA      = ws + o; o += (size_t)50 * 256 * 196;
    float* mB      = ws + o; o += (size_t)50 * 256 * 196;
    float* dc      = ws + o; o += (size_t)50 * 256 * 784;
    (void)ws_size; (void)in_sizes; (void)n_in; (void)out_size;

    float* out0 = (float*)d_out;            // (2,735,2)
    float* out1 = out0 + 2940;              // (2,735,4)
    float* out2 = out0 + 8820;              // (2,250,20) logits
    float* out3 = out0 + 18820;             // (2,250,20,4)
    float* out4 = out0 + 58820;             // (2,250,20,28,28)

    const int FS[5]  = {104, 52, 26, 13, 7};
    const int STR[5] = {4, 8, 16, 32, 64};

    auto blocks2 = [](int P, int OC) {
        int tiles = ((P + 31) / 32) * ((OC + 31) / 32);
        return (tiles + 3) / 4;
    };

    // ---------------- RPN + proposals, per level ----------------
    for (int lvl = 0; lvl < 5; ++lvl) {
        int H = FS[lvl];
        int P = 2 * H * H;
        conv_wmma_kernel<3><<<blocks2(P, 512), 128, 0, stream>>>(
            feat[lvl], rpn_ws, rpn_bs, nullptr, nullptr, nullptr, nullptr,
            h_rpn, 2, 256, H, H, 512, 1, /*epi=*/1);
        conv_wmma_kernel<1><<<blocks2(P, 30), 128, 0, stream>>>(
            h_rpn, rpn_wc, rpn_bc, nullptr, nullptr, nullptr, nullptr,
            clsBuf, 2, 512, H, H, 30, 0, /*epi=*/0);
        conv_wmma_kernel<1><<<blocks2(P, 60), 128, 0, stream>>>(
            h_rpn, rpn_wb, rpn_bb, nullptr, nullptr, nullptr, nullptr,
            bboxBuf, 2, 512, H, H, 60, 0, /*epi=*/0);
        proposals_kernel<<<2, 256, 0, stream>>>(
            clsBuf, bboxBuf, scoreB, boxes, lvl, H, (float)STR[lvl]);
    }
    // level-4 raw heads -> out0/out1 (clsBuf/bboxBuf still hold level 4)
    rpn_head_transpose_kernel<<<(2940 + 255) / 256, 256, 0, stream>>>(clsBuf, out0, 2, 7);
    rpn_head_transpose_kernel<<<(5880 + 255) / 256, 256, 0, stream>>>(bboxBuf, out1, 4, 7);

    // ---------------- ROI align ----------------
    {
        int tot7  = 2 * 250 * 256 * 49;
        int tot14 = 2 * 250 * 256 * 196;
        roi_align_kernel<<<(tot7 + 255) / 256, 256, 0, stream>>>(
            feat[0], feat[1], feat[2], feat[3], feat[4], boxes, pf, 7);
        roi_align_kernel<<<(tot14 + 255) / 256, 256, 0, stream>>>(
            feat[0], feat[1], feat[2], feat[3], feat[4], boxes, mfb, 14);
    }

    // ---------------- Classifier ----------------
    gemm_wmma_kernel<<<blocks2(500, 1024), 128, 0, stream>>>(
        pf, cls_w1, cls_b1, bn1_g, bn1_b, bn1_m, bn1_v, h1, 500, 1024, 12544, /*epi=*/4);
    gemm_wmma_kernel<<<blocks2(500, 1024), 128, 0, stream>>>(
        h1, cls_w2, cls_b2, bn2_g, bn2_b, bn2_m, bn2_v, h2, 500, 1024, 1024, /*epi=*/2);
    gemm_wmma_kernel<<<blocks2(500, 20), 128, 0, stream>>>(
        h2, cls_wfc, cls_bfc, nullptr, nullptr, nullptr, nullptr, out2, 500, 20, 1024, 0);
    gemm_wmma_kernel<<<blocks2(500, 80), 128, 0, stream>>>(
        h2, cls_wbb, cls_bbb, nullptr, nullptr, nullptr, nullptr, out3, 500, 80, 1024, 0);

    // ---------------- Mask branch, chunks of 50 ROIs ----------------
    for (int ch = 0; ch < 10; ++ch) {
        const float* src = mfb + (size_t)ch * 50 * 256 * 196;
        int P14 = 50 * 196, P28 = 50 * 784;
        conv_wmma_kernel<3><<<blocks2(P14, 256), 128, 0, stream>>>(
            src, m_w[0], m_b[0], m_bn_g[0], m_bn_b[0], m_bn_m[0], m_bn_v[0],
            mA, 50, 256, 14, 14, 256, 1, 2);
        conv_wmma_kernel<3><<<blocks2(P14, 256), 128, 0, stream>>>(
            mA, m_w[1], m_b[1], m_bn_g[1], m_bn_b[1], m_bn_m[1], m_bn_v[1],
            mB, 50, 256, 14, 14, 256, 1, 2);
        conv_wmma_kernel<3><<<blocks2(P14, 256), 128, 0, stream>>>(
            mB, m_w[2], m_b[2], m_bn_g[2], m_bn_b[2], m_bn_m[2], m_bn_v[2],
            mA, 50, 256, 14, 14, 256, 1, 2);
        conv_wmma_kernel<3><<<blocks2(P14, 256), 128, 0, stream>>>(
            mA, m_w[3], m_b[3], m_bn_g[3], m_bn_b[3], m_bn_m[3], m_bn_v[3],
            mB, 50, 256, 14, 14, 256, 1, 2);
        {
            int tiles = ((P14 + 31) / 32) * 8 * 4;
            deconv_wmma_kernel<<<(tiles + 3) / 4, 128, 0, stream>>>(mB, m_wd, m_bd, dc, 50);
        }
        conv_wmma_kernel<1><<<blocks2(P28, 20), 128, 0, stream>>>(
            dc, m_wf, m_bf, nullptr, nullptr, nullptr, nullptr,
            out4 + (size_t)ch * 50 * 20 * 784, 50, 256, 28, 28, 20, 0, /*epi=*/3);
    }
}
